// RBMBernoulli_10900626997656
// MI455X (gfx1250) — matrix operand order repaired and checked by reading, not compiled, not run on hardware
//
#include <hip/hip_runtime.h>
#include <hip/hip_bf16.h>

// ---------------------------------------------------------------------------
// RBM Bernoulli CD-k, exploiting the rank-2 carry:
//   sample 0 : dense W (12288x4096 f32) -> 8 memory-bound GEMVs (WMMA-based)
//   samples 1..15 : W is rank-2 LR*(p q^T - r s^T) -> dots + elementwise
// Output: reconstruction v of sample 15 (12288 f32).
// ---------------------------------------------------------------------------

#define HID   4096
#define FLATN 12288
#define NSAMP 16
#define KSTEP 4
#define LR    0.01f

#define NS_T 8                       // reduction slices for h = sigma(W^T v)
#define SLICE_T (FLATN / NS_T)       // 1536
#define NS_V 4                       // reduction slices for v = sigma(W h)
#define SLICE_V (HID / NS_V)         // 1024

typedef __attribute__((ext_vector_type(16))) _Float16 v16h;
typedef __attribute__((ext_vector_type(8)))  float    v8f;

__device__ __forceinline__ float sigmoidf(float x) {
    return 1.0f / (1.0f + __expf(-x));
}

// ---------------------------------------------------------------------------
// partial[slice*HID + j] = sum_{i in slice} W[i,j] * v[i]
// One wave per (16-column tile, slice). A = broadcast v chunk (16x32),
// B = W tile (32x16), D accumulated over the i dimension via WMMA.
// ---------------------------------------------------------------------------
__global__ void gemvT_wmma_kernel(const float* __restrict__ W,
                                  const float* __restrict__ v,
                                  float* __restrict__ part) {
    const int tile  = blockIdx.x;       // 0..HID/16-1
    const int slice = blockIdx.y;       // 0..NS_T-1
    const int lane  = threadIdx.x;      // 0..31
    const int j0    = tile * 16;
    const int n     = lane & 15;
    const int koffB = (lane < 16) ? 0 : 16;  // B: lanes 0-15 hold K 0..15
    const int koffA = (lane < 16) ? 0 : 16;  // A: lanes 0-15 hold K 0..15

    v8f c = {};
    const int iBeg = slice * SLICE_T;
    const int iEnd = iBeg + SLICE_T;
    for (int i0 = iBeg; i0 < iEnd; i0 += 32) {
        v16h af, bf;
#pragma unroll
        for (int t = 0; t < 8; ++t) {
            af[t]     = (_Float16)v[i0 + koffA + t];
            af[8 + t] = (_Float16)v[i0 + 8 + koffA + t];
        }
        const float* wp = W + (size_t)(i0 + koffB) * HID + (j0 + n);
        __builtin_prefetch(wp + (size_t)32 * HID, 0, 1);   // next i-tile
#pragma unroll
        for (int t = 0; t < 16; ++t)
            bf[t] = (_Float16)wp[(size_t)t * HID];
        c = __builtin_amdgcn_wmma_f32_16x16x32_f16(false, af, false, bf,
                                                   (short)0, c, false, false);
    }
    if (lane < 16) part[slice * HID + j0 + lane] = c[0];   // row M=0 of D
}

// ---------------------------------------------------------------------------
// partial[slice*FLATN + i] = sum_{j in slice} W[i,j] * h[j]
// One wave per (16-row tile, slice). B tile rows are contiguous -> float4.
// ---------------------------------------------------------------------------
__global__ void gemv_wmma_kernel(const float* __restrict__ W,
                                 const float* __restrict__ h,
                                 float* __restrict__ part) {
    const int tile  = blockIdx.x;       // 0..FLATN/16-1
    const int slice = blockIdx.y;       // 0..NS_V-1
    const int lane  = threadIdx.x;
    const int i0    = tile * 16;
    const int n     = lane & 15;
    const int koffB = (lane < 16) ? 0 : 16;
    const int koffA = (lane < 16) ? 0 : 16;

    v8f c = {};
    const int jBeg = slice * SLICE_V;
    const int jEnd = jBeg + SLICE_V;
    const float* wrow = W + (size_t)(i0 + n) * HID;
    for (int j0 = jBeg; j0 < jEnd; j0 += 32) {
        v16h af, bf;
#pragma unroll
        for (int t = 0; t < 8; ++t) {
            af[t]     = (_Float16)h[j0 + koffA + t];
            af[8 + t] = (_Float16)h[j0 + 8 + koffA + t];
        }
        const float4* wp = (const float4*)(wrow + j0 + koffB);  // 64B aligned
#pragma unroll
        for (int t4 = 0; t4 < 4; ++t4) {
            float4 w4 = wp[t4];
            bf[t4 * 4 + 0] = (_Float16)w4.x;
            bf[t4 * 4 + 1] = (_Float16)w4.y;
            bf[t4 * 4 + 2] = (_Float16)w4.z;
            bf[t4 * 4 + 3] = (_Float16)w4.w;
        }
        c = __builtin_amdgcn_wmma_f32_16x16x32_f16(false, af, false, bf,
                                                   (short)0, c, false, false);
    }
    if (lane < 16) part[slice * FLATN + i0 + lane] = c[0];
}

// out[i] = sigmoid(bias[i] + sum_s part[s*n + i])
__global__ void reduce_sigmoid_kernel(const float* __restrict__ part,
                                      const float* __restrict__ bias,
                                      float* __restrict__ out, int n, int nslice) {
    int i = blockIdx.x * blockDim.x + threadIdx.x;
    if (i >= n) return;
    float s = bias[i];
    for (int sl = 0; sl < nslice; ++sl) s += part[sl * n + i];
    out[i] = sigmoidf(s);
}

// out[0] = x . y1 ; out[1] = x . y2   (single block, deterministic reduction)
__global__ void dots2_kernel(const float* __restrict__ x,
                             const float* __restrict__ y1,
                             const float* __restrict__ y2,
                             int n, float* __restrict__ out) {
    __shared__ float s1[1024];
    __shared__ float s2[1024];
    int t = threadIdx.x;
    float d1 = 0.f, d2 = 0.f;
    for (int i = t; i < n; i += 1024) {
        float xv = x[i];
        d1 += xv * y1[i];
        d2 += xv * y2[i];
    }
    s1[t] = d1; s2[t] = d2;
    __syncthreads();
    for (int st = 512; st > 0; st >>= 1) {
        if (t < st) { s1[t] += s1[t + st]; s2[t] += s2[t + st]; }
        __syncthreads();
    }
    if (t == 0) { out[0] = s1[0]; out[1] = s2[0]; }
}

// out[i] = sigmoid(bias[i] + LR*(f1[i]*sc[0] - f2[i]*sc[1]))
__global__ void rank2_sigmoid_kernel(const float* __restrict__ bias,
                                     const float* __restrict__ f1,
                                     const float* __restrict__ f2,
                                     const float* __restrict__ sc,
                                     float* __restrict__ out, int n) {
    int i = blockIdx.x * blockDim.x + threadIdx.x;
    if (i >= n) return;
    float x = bias[i] + LR * (f1[i] * sc[0] - f2[i] * sc[1]);
    out[i] = sigmoidf(x);
}

// out[i] = LR*(u[i] - w[i])
__global__ void carry_kernel(const float* __restrict__ u,
                             const float* __restrict__ w,
                             float* __restrict__ out, int n) {
    int i = blockIdx.x * blockDim.x + threadIdx.x;
    if (i < n) out[i] = LR * (u[i] - w[i]);
}

__global__ void copy_kernel(const float* __restrict__ src,
                            float* __restrict__ dst, int n) {
    int i = blockIdx.x * blockDim.x + threadIdx.x;
    if (i < n) dst[i] = src[i];
}

// ---------------------------------------------------------------------------
static inline int cdiv(int a, int b) { return (a + b - 1) / b; }

extern "C" void kernel_launch(void* const* d_in, const int* in_sizes, int n_in,
                              void* d_out, int out_size, void* d_ws, size_t ws_size,
                              hipStream_t stream) {
    const float* inputs = (const float*)d_in[0];   // (16, 64, 64, 3) row-major
    const float* W      = (const float*)d_in[1];   // (FLATN, HID)
    const float* a0     = (const float*)d_in[2];   // (FLATN,1) zeros
    const float* b0     = (const float*)d_in[3];   // (HID,1)   zeros
    float* out = (float*)d_out;

    // Workspace layout (floats). Total ~410 KB.
    float* ws   = (float*)d_ws;
    float* a    = ws;                 // FLATN
    float* b    = a + FLATN;          // HID
    float* vA   = b + HID;            // FLATN
    float* vB   = vA + FLATN;         // FLATN
    float* hA   = vB + FLATN;         // HID
    float* hB   = hA + HID;           // HID
    float* qA   = hB + HID;           // HID
    float* qB   = qA + HID;           // HID
    float* part = qB + HID;           // max(NS_T*HID, NS_V*FLATN) = 49152
    float* sc   = part + NS_V * FLATN;// 8 scalars

    const int BLK = 256;
    const int gF  = cdiv(FLATN, BLK);
    const int gH  = cdiv(HID, BLK);

    float *vcur = vA, *rfac = vB;     // current v / previous-sample final v
    float *hcur = hA, *sfac = hB;     // current h / previous-sample final h
    float *qfac = qA, *qnext = qB;    // previous-sample h_init / current h_init
    const float* P = inputs;          // p factor = v_init of previous sample

    // ---- carry init (a = b = 0 per reference) ----
    copy_kernel<<<gF, BLK, 0, stream>>>(a0, a, FLATN);
    copy_kernel<<<gH, BLK, 0, stream>>>(b0, b, HID);

    // ---- sample 0: dense W Gibbs chain ----
    copy_kernel<<<gF, BLK, 0, stream>>>(inputs, vcur, FLATN);
    for (int k = 0; k < KSTEP; ++k) {
        gemvT_wmma_kernel<<<dim3(HID / 16, NS_T), 32, 0, stream>>>(W, vcur, part);
        reduce_sigmoid_kernel<<<gH, BLK, 0, stream>>>(part, b, hcur, HID, NS_T);
        if (k == 0)  // h at k==0 with v==v_init IS h_init
            copy_kernel<<<gH, BLK, 0, stream>>>(hcur, qnext, HID);
        gemv_wmma_kernel<<<dim3(FLATN / 16, NS_V), 32, 0, stream>>>(W, hcur, part);
        reduce_sigmoid_kernel<<<gF, BLK, 0, stream>>>(part, a, vcur, FLATN, NS_V);
    }
    // carry after sample 0: a=LR*(p-v), b=LR*(q-h); W becomes rank-2 (p,q,r,s)
    carry_kernel<<<gF, BLK, 0, stream>>>(P, vcur, a, FLATN);
    carry_kernel<<<gH, BLK, 0, stream>>>(qnext, hcur, b, HID);
    { float* t = vcur; vcur = rfac; rfac = t; }
    { float* t = hcur; hcur = sfac; sfac = t; }
    { float* t = qfac; qfac = qnext; qnext = t; }

    // ---- samples 1..15: rank-2 W = LR*(p q^T - r s^T) ----
    for (int n = 1; n < NSAMP; ++n) {
        const float* vinit = inputs + (size_t)n * FLATN;
        copy_kernel<<<gF, BLK, 0, stream>>>(vinit, vcur, FLATN);
        for (int k = 0; k < KSTEP; ++k) {
            // h = sigmoid(b + W^T v) = sigmoid(b + LR*(q (p.v) - s (r.v)))
            dots2_kernel<<<1, 1024, 0, stream>>>(vcur, P, rfac, FLATN, sc);
            rank2_sigmoid_kernel<<<gH, BLK, 0, stream>>>(b, qfac, sfac, sc, hcur, HID);
            if (k == 0)
                copy_kernel<<<gH, BLK, 0, stream>>>(hcur, qnext, HID);
            // v = sigmoid(a + W h) = sigmoid(a + LR*(p (q.h) - r (s.h)))
            dots2_kernel<<<1, 1024, 0, stream>>>(hcur, qfac, sfac, HID, sc + 2);
            rank2_sigmoid_kernel<<<gF, BLK, 0, stream>>>(a, P, rfac, sc + 2, vcur, FLATN);
        }
        if (n == NSAMP - 1) {
            copy_kernel<<<gF, BLK, 0, stream>>>(vcur, out, FLATN);
            break;  // carry of the last sample is never observed
        }
        carry_kernel<<<gF, BLK, 0, stream>>>(vinit, vcur, a, FLATN);
        carry_kernel<<<gH, BLK, 0, stream>>>(qnext, hcur, b, HID);
        { float* t = vcur; vcur = rfac; rfac = t; }
        { float* t = hcur; hcur = sfac; sfac = t; }
        { float* t = qfac; qfac = qnext; qnext = t; }
        P = vinit;
    }

    (void)in_sizes; (void)n_in; (void)out_size; (void)ws_size;
}
